// AdaptivePatchEmbedding_54030688584036
// MI455X (gfx1250) — matrix operand order repaired and checked by reading, not compiled, not run on hardware
//
#include <hip/hip_runtime.h>
#include <hip/hip_bf16.h>

// ---------------------------------------------------------------------------
// AdaptivePatchEmbedding for MI455X (gfx1250, wave32, WMMA)
// ---------------------------------------------------------------------------
typedef _Float16 half_t;
typedef __attribute__((ext_vector_type(16))) _Float16 v16h;
typedef __attribute__((ext_vector_type(8)))  _Float16 v8h;
typedef __attribute__((ext_vector_type(8)))  float    v8f;

#define NB   32
#define NC   22
#define NT   5000
#define NOC  40
#define NP   100
#define NEMB 64
#define NFC  120

__device__ __forceinline__ unsigned f2key(float f) {
    unsigned u = __float_as_uint(f);
    return (u & 0x80000000u) ? ~u : (u | 0x80000000u);
}
__device__ __forceinline__ float key2f(unsigned k) {
    unsigned u = (k & 0x80000000u) ? (k & 0x7fffffffu) : ~k;
    return __uint_as_float(u);
}

// ---------------------------------------------------------------------------
// Kernel 1: STFT power -> comb[b,c,f]  (only the 11 used bins, DFT w/ LDS
// twiddle table; reflect padding; periodic hann)
// ---------------------------------------------------------------------------
__global__ __launch_bounds__(256) void stft_comb_kernel(
    const float* __restrict__ x, float* __restrict__ comb)
{
    __shared__ float cosT[128], sinT[128];
    int tid = threadIdx.x;
    if (tid < 128) {
        float ang = 6.283185307179586f * (float)tid / 128.0f;
        cosT[tid] = cosf(ang);
        sinT[tid] = sinf(ang);
    }
    __syncthreads();
    int id = blockIdx.x * 256 + tid;        // 550*256 == 140800 exactly
    if (id >= NB * NC * 200) return;
    int f  = id % 200;
    int bc = id / 200;
    const float* xr = x + (long)bc * NT;

    float re[11], im[11];
#pragma unroll
    for (int k = 0; k < 11; ++k) { re[k] = 0.f; im[k] = 0.f; }

    int base = f * 25 - 64;
    for (int n = 0; n < 128; ++n) {
        int tt = base + n;
        if (tt < 0)   tt = -tt;                // reflect left
        if (tt >= NT) tt = 2 * NT - 2 - tt;    // reflect right
        float v = xr[tt] * (0.5f - 0.5f * cosT[n]);
        int j = (5 * n) & 127;                 // bin-5 phase, +n per bin
#pragma unroll
        for (int k = 0; k < 11; ++k) {
            re[k] += v * cosT[j];
            im[k] -= v * sinT[j];
            j = (j + n) & 127;
        }
    }
    float p[11];
#pragma unroll
    for (int k = 0; k < 11; ++k) p[k] = re[k] * re[k] + im[k] * im[k];
    float mu = 0.5f * (p[0] + p[1]);
    float be = 0.f;
#pragma unroll
    for (int k = 2; k < 11; ++k) be += p[k];
    comb[id] = mu + be * (1.0f / 9.0f);
}

// ---------------------------------------------------------------------------
// Kernel 2: fused conv weights, K layout q = c*(ks+1)+k (even channel runs;
// k==ks slot is a zero pad so paired B loads never straddle a channel).
// W_eff[o, c*KSP+k] = sum_i w2[o,i,c]*w1[i,k];  padded [48, Kpad] f16.
// biasEff[o] = b2[o] + sum_{i,c} w2[o,i,c]*b1[i]
// ---------------------------------------------------------------------------
__global__ void weff_prep_kernel(
    const float* __restrict__ w1, const float* __restrict__ w2,
    const float* __restrict__ b1, const float* __restrict__ b2,
    half_t* __restrict__ weff, float* __restrict__ biasEff, int ks, int Kpad)
{
    int ksp = ks + 1;
    int idx = blockIdx.x * blockDim.x + threadIdx.x;
    int total = 48 * Kpad;
    if (idx < total) {
        int o = idx / Kpad, q = idx % Kpad;
        int c = q / ksp, k = q - c * ksp;
        float acc = 0.f;
        if (o < NOC && c < NC && k < ks) {
            for (int i = 0; i < NOC; ++i)
                acc += w2[(o * NOC + i) * NC + c] * w1[i * ks + k];
        }
        weff[o * Kpad + q] = (half_t)acc;
    }
    if (idx < NOC) {
        float acc = b2[idx];
        for (int i = 0; i < NOC; ++i) {
            float s = 0.f;
            for (int c = 0; c < NC; ++c) s += w2[(idx * NOC + i) * NC + c];
            acc += s * b1[i];
        }
        biasEff[idx] = acc;
    } else if (idx < 48) {
        biasEff[idx] = 0.f;
    }
}

// ---------------------------------------------------------------------------
// Kernel 3: activity detector + linear interp + adaptive grid + ends
// ---------------------------------------------------------------------------
__global__ __launch_bounds__(256) void activity_grid_kernel(
    const float* __restrict__ comb,
    const float* __restrict__ aw1, const float* __restrict__ ab1,
    const float* __restrict__ ag,  const float* __restrict__ abt,
    const float* __restrict__ am,  const float* __restrict__ av,
    const float* __restrict__ aw2, const float* __restrict__ ab2,
    float* __restrict__ grid_out, float* __restrict__ acti_out,
    int* __restrict__ ends)
{
    __shared__ float cs[NC * 200];
    __shared__ float a1[11 * 200];
    __shared__ float act[200];
    __shared__ float wv[NP];
    __shared__ float gs[NP];
    __shared__ int   gi[NP];

    int b = blockIdx.x, tid = threadIdx.x;
    for (int i = tid; i < NC * 200; i += 256) cs[i] = comb[b * NC * 200 + i];
    __syncthreads();

    for (int i = tid; i < 11 * 200; i += 256) {
        int o = i / 200, t = i % 200;
        float acc = ab1[o];
        for (int ci = 0; ci < NC; ++ci) {
            const float* row = cs + ci * 200;
            const float* w   = aw1 + (o * NC + ci) * 3;
            float s = row[t] * w[1];
            if (t > 0)   s += row[t - 1] * w[0];
            if (t < 199) s += row[t + 1] * w[2];
            acc += s;
        }
        float y = (acc - am[o]) * (ag[o] * rsqrtf(av[o] + 1e-5f)) + abt[o];
        a1[i] = y > 0.f ? y : 0.f;
    }
    __syncthreads();

    if (tid < 200) {
        float acc = ab2[0];
        for (int o = 0; o < 11; ++o) acc += aw2[o] * a1[o * 200 + tid];
        act[tid] = 1.f / (1.f + expf(-acc));
    }
    __syncthreads();

    if (tid < NP) {
        float ai = 0.5f * (act[2 * tid] + act[2 * tid + 1]); // pos = 2p+0.5
        acti_out[b * NP + tid] = ai;
        wv[tid] = 1.f / (ai + 1e-6f);
    }
    __syncthreads();

    if (tid == 0) {
        float sumw = 0.f;
        for (int p = 0; p < NP; ++p) sumw += wv[p];
        float sum2 = 0.f;
        for (int p = 0; p < NP; ++p) {
            float g = wv[p] / sumw * (float)NT;
            g = fminf(fmaxf(g, 10.f), 100.f);
            gs[p] = g; sum2 += g;
        }
        float sc = (float)NT / sum2;
        int sum3 = 0;
        for (int p = 0; p < NP; ++p) {
            int g = (int)rintf(gs[p] * sc);     // round-half-even
            gi[p] = g; sum3 += g;
        }
        gi[NP - 1] += NT - sum3;
        int sum4 = 0;
        for (int p = 0; p < NP; ++p) {
            int g = min(max(gi[p], 10), 100);
            gi[p] = g; sum4 += g;
        }
        gi[NP - 1] += NT - sum4;
        int cum = 0;
        for (int p = 0; p < NP; ++p) {
            int g = max(gi[p], 10);
            grid_out[b * NP + p] = (float)g;
            cum += g;
            ends[b * NP + p] = cum;
        }
    }
}

// ---------------------------------------------------------------------------
// Kernel 4: fused conv via WMMA. One wave owns all 3 M-tiles (M=48) of its
// 16-wide t-tile; block = 6 waves = 96 t. x tile staged in LDS as overlapping
// f16 pairs: xs2[c][e] = pack(x[e], x[e+1]) so each 2 K-elements of the B
// fragment is ONE aligned ds_load_b32 in final bit layout (8 DS loads per
// chunk feeding 3 WMMAs). Channel runs padded to even length (KSP=ks+1) so
// pairs never straddle a run. Epilogue: fused BN+bias, ELU, two-level pooling
// (LDS ds_add_f32/ds_max_u32 slots, then one global atomic per seg/channel).
// ---------------------------------------------------------------------------
__device__ __forceinline__ int seg_search(const int* ends_s, int t) {
    int lo = 0, hi = NP;        // searchsorted(ends, t, 'right')
    while (lo < hi) { int mid = (lo + hi) >> 1; if (ends_s[mid] <= t) lo = mid + 1; else hi = mid; }
    return lo;
}

template <int KS>
__global__ __launch_bounds__(192) void conv_branch_kernel(
    const float* __restrict__ x, const half_t* __restrict__ weff,
    const float* __restrict__ biasEff,
    const float* __restrict__ bng, const float* __restrict__ bnb,
    const float* __restrict__ bnm, const float* __restrict__ bnv,
    const int* __restrict__ ends,
    float* __restrict__ sumbuf, unsigned* __restrict__ maxbuf,
    int ocOff)
{
    constexpr int KSP    = KS + 1;                    // even channel run
    constexpr int KPAD   = (NC * KSP + 31) & ~31;
    constexpr int NCH    = KPAD / 32;
    constexpr int H0     = (KS - 1) / 2;
    constexpr int WID    = 96 + KS - 1;               // pair entries per row
    constexpr int PITCH2 = 160;                       // uint entries per row

    __shared__ unsigned xs2[NC * PITCH2];             // paired f16 tile
    __shared__ int      ends_s[NP];
    __shared__ float    sc_s[48], sf_s[48];
    __shared__ float    seg_sum[16 * NOC];
    __shared__ unsigned seg_max[16 * NOC];

    int b  = blockIdx.y;
    int t0 = blockIdx.x * 96;
    int tid = threadIdx.x;

    // ---- block init: ends, fused BN constants, pool slots, paired x tile --
    for (int i = tid; i < NP; i += 192) ends_s[i] = ends[b * NP + i];
    for (int i = tid; i < 16 * NOC; i += 192) { seg_sum[i] = 0.f; seg_max[i] = 0u; }
    if (tid < 48) {
        if (tid < NOC) {
            float sc = bng[tid] * rsqrtf(bnv[tid] + 1e-5f);
            sc_s[tid] = sc;
            sf_s[tid] = bnb[tid] - bnm[tid] * sc + biasEff[tid] * sc;
        } else { sc_s[tid] = 0.f; sf_s[tid] = 0.f; }
    }
    const float* xb = x + (long)b * NC * NT;
    for (int i = tid; i < NC * WID; i += 192) {
        int c = i / WID, e = i % WID;
        int t = t0 - H0 + e;
        float v0 = (t >= 0 && t < NT)         ? xb[c * NT + t]     : 0.f;
        float v1 = (t + 1 >= 0 && t + 1 < NT) ? xb[c * NT + t + 1] : 0.f;
        union { half_t h[2]; unsigned u; } pk;
        pk.h[0] = (half_t)v0; pk.h[1] = (half_t)v1;
        xs2[c * PITCH2 + e] = pk.u;
    }
    __syncthreads();

    int wave = tid >> 5, lane = tid & 31;
    int laneM  = lane & 15;
    int hiHalf = lane >> 4;
    int kbA = hiHalf * 16;      // A frag: K = kc*32 + kbA + h
    int kbB = hiHalf * 16;      // B frag: K = kc*32 + kbB + h
    int tcol = wave * 16 + laneM;     // 0..95

    v8f acc[3] = {v8f{}, v8f{}, v8f{}};
    const half_t* wrow0 = weff + (long)laneM * KPAD;   // rows laneM, 16+, 32+

    for (int kc = 0; kc < NCH; ++kc) {
        // ---- A fragments for the 3 M tiles (16B vector loads) ----
        v16h A[3];
#pragma unroll
        for (int mt = 0; mt < 3; ++mt) {
            const half_t* pa = wrow0 + (long)mt * 16 * KPAD + kc * 32 + kbA;
            v8h a0 = *(const v8h*)pa;
            v8h a1 = *(const v8h*)(pa + 8);
#pragma unroll
            for (int j = 0; j < 8; ++j) { A[mt][j] = a0[j]; A[mt][8 + j] = a1[j]; }
        }
        // ---- B fragment: 8 paired dword LDS loads, no repacking ----
        union { v16h h; unsigned u[8]; } bf;
#pragma unroll
        for (int j = 0; j < 8; ++j) {
            int q  = kc * 32 + kbB + 2 * j;   // even; run-aligned (KSP even)
            int c  = q / KSP;
            int kk = q - c * KSP;
            if (c >= NC) c = NC - 1;          // K padding: A is zero there
            bf.u[j] = xs2[c * PITCH2 + tcol + kk];
        }
#pragma unroll
        for (int mt = 0; mt < 3; ++mt)
            acc[mt] = __builtin_amdgcn_wmma_f32_16x16x32_f16(
                          false, A[mt], false, bf.h, (short)0, acc[mt], false, false);
    }

    // ---- epilogue: ELU(BN(conv+bias)) -> LDS segment pooling ----
    int seg0 = seg_search(ends_s, t0);
    int t = t0 + tcol;
    if (t < NT) {
        int sl = seg_search(ends_s, t) - seg0;
        sl = min(max(sl, 0), 15);
        int slotBase = sl * NOC;
#pragma unroll
        for (int mt = 0; mt < 3; ++mt) {
#pragma unroll
            for (int vv = 0; vv < 8; ++vv) {
                int o = mt * 16 + vv + hiHalf * 8;
                if (o < NOC) {
                    float y = acc[mt][vv] * sc_s[o] + sf_s[o];
                    float e = y > 0.f ? y : (expf(y) - 1.f);     // ELU
                    atomicAdd(&seg_sum[slotBase + o], e);        // ds_add_f32
                    atomicMax(&seg_max[slotBase + o], f2key(e)); // ds_max_u32
                }
            }
        }
    }
    __syncthreads();

    // ---- flush <=16 local segments to global accumulators ----
    int tLast = min(t0 + 95, NT - 1);
    int nseg  = seg_search(ends_s, tLast) - seg0 + 1;
    if (nseg > 16) nseg = 16;
    for (int i = tid; i < nseg * NOC; i += 192) {
        int sl = i / NOC, o = i - sl * NOC;
        int rowBase = (b * (NP + 1) + seg0 + sl) * NFC + ocOff + o;
        atomicAdd(&sumbuf[rowBase], seg_sum[i]);
        atomicMax(&maxbuf[rowBase], seg_max[i]);
    }
}

// ---------------------------------------------------------------------------
// Kernel 5: finalize pooled = avg + max with validity masks
// ---------------------------------------------------------------------------
__global__ void pool_fin_kernel(
    const float* __restrict__ sumbuf, const unsigned* __restrict__ maxbuf,
    const int* __restrict__ ends, float* __restrict__ pooled)
{
    int idx = blockIdx.x * 256 + threadIdx.x;
    if (idx >= NB * NP * NFC) return;
    int o = idx % NFC; int bp = idx / NFC;
    int p = bp % NP;   int b  = bp / NP;
    int ep = ends[b * NP + p];
    int em = p ? ends[b * NP + p - 1] : 0;
    int cnt = min(ep, NT) - min(em, NT);
    if (cnt < 0) cnt = 0;
    int src = (b * (NP + 1) + p) * NFC + o;
    float avg = sumbuf[src] / fmaxf((float)cnt, 1.f);
    float mx  = cnt > 0 ? key2f(maxbuf[src]) : 0.f;
    bool valid = (ep <= NT) && (em < ep);
    pooled[idx] = valid ? (avg + mx) : 0.f;
}

// ---------------------------------------------------------------------------
// Kernel 6: projection emb[n,o] = proj_w[o,:] . pooled[n,:] + pb  via WMMA
// M=64 (4 tiles), K=120->128 (4 chunks), N=3200 (200 tiles). 8 waves/block.
// ---------------------------------------------------------------------------
__global__ __launch_bounds__(256) void proj_wmma_kernel(
    const float* __restrict__ pooled, const float* __restrict__ pw,
    const float* __restrict__ pb, float* __restrict__ out)
{
    int tid = threadIdx.x;
    int gw  = blockIdx.x * 8 + (tid >> 5);   // 0..799
    int lane = tid & 31;
    int m_tile = gw & 3;
    int n_tile = gw >> 2;
    int laneM = lane & 15, hiHalf = lane >> 4;
    int n = n_tile * 16 + laneM;             // < 3200
    int kbA = hiHalf * 8, kbB = hiHalf * 16;

    v8f acc = {};
    const float* prow = pw + (m_tile * 16 + laneM) * NFC;
    const float* brow = pooled + (long)n * NFC;
#pragma unroll
    for (int kc = 0; kc < 4; ++kc) {
        v16h A, Bf;
#pragma unroll
        for (int h = 0; h < 16; ++h) {
            int k = kc * 32 + (h >> 3) * 16 + kbA + (h & 7);
            A[h] = (half_t)(k < NFC ? prow[k] : 0.f);
        }
#pragma unroll
        for (int h = 0; h < 16; ++h) {
            int k = kc * 32 + (h >> 3) * 16 + kbA + (h & 7);
            Bf[h] = (half_t)(k < NFC ? brow[k] : 0.f);
        }
        acc = __builtin_amdgcn_wmma_f32_16x16x32_f16(
                  false, A, false, Bf, (short)0, acc, false, false);
    }
#pragma unroll
    for (int vv = 0; vv < 8; ++vv) {
        int o = m_tile * 16 + vv + hiHalf * 8;
        out[(long)n * NEMB + o] = acc[vv] + pb[o];
    }
}

// ---------------------------------------------------------------------------
// Host launcher
// ---------------------------------------------------------------------------
extern "C" void kernel_launch(void* const* d_in, const int* in_sizes, int n_in,
                              void* d_out, int out_size, void* d_ws, size_t ws_size,
                              hipStream_t stream)
{
    (void)in_sizes; (void)n_in; (void)out_size; (void)ws_size;
    const float* x = (const float*)d_in[0];
    const float* ms_w1[3]; const float* ms_b1[3]; const float* ms_w2[3]; const float* ms_b2[3];
    const float* ms_g[3];  const float* ms_bt[3]; const float* ms_m[3];  const float* ms_v[3];
    for (int s = 0; s < 3; ++s) {
        int base = 1 + s * 8;
        ms_w1[s] = (const float*)d_in[base + 0];
        ms_b1[s] = (const float*)d_in[base + 1];
        ms_w2[s] = (const float*)d_in[base + 2];
        ms_b2[s] = (const float*)d_in[base + 3];
        ms_g[s]  = (const float*)d_in[base + 4];
        ms_bt[s] = (const float*)d_in[base + 5];
        ms_m[s]  = (const float*)d_in[base + 6];
        ms_v[s]  = (const float*)d_in[base + 7];
    }
    const float* a_w1 = (const float*)d_in[25];
    const float* a_b1 = (const float*)d_in[26];
    const float* a_g  = (const float*)d_in[27];
    const float* a_bt = (const float*)d_in[28];
    const float* a_m  = (const float*)d_in[29];
    const float* a_v  = (const float*)d_in[30];
    const float* a_w2 = (const float*)d_in[31];
    const float* a_b2 = (const float*)d_in[32];
    const float* proj_w = (const float*)d_in[33];
    const float* proj_b = (const float*)d_in[34];

    // K layout uses KSP = ks+1: Kpad = ceil(22*(ks+1)/32)*32
    const int KpadArr[3] = {224, 576, 1120};   // ks = 9 / 25 / 49
    const int ksArr[3]   = {9, 25, 49};

    // workspace layout (256B aligned)
    char* base = (char*)d_ws;
    size_t off = 0;
    auto align256 = [](size_t v) { return (v + 255) & ~(size_t)255; };
    size_t comb_off = off;               off += align256((size_t)NB * NC * 200 * 4);
    size_t weff_off[3];
    for (int s = 0; s < 3; ++s) { weff_off[s] = off; off += align256((size_t)48 * KpadArr[s] * 2); }
    size_t bias_off = off;               off += align256(3 * 64 * 4);
    size_t ends_off = off;               off += align256((size_t)NB * NP * 4);
    size_t sum_off  = off;               off += align256((size_t)NB * (NP + 1) * NFC * 4);
    size_t max_off  = off;               off += align256((size_t)NB * (NP + 1) * NFC * 4);
    size_t pool_off = off;               off += align256((size_t)NB * NP * NFC * 4);

    float*    comb   = (float*)(base + comb_off);
    float*    biasE  = (float*)(base + bias_off);
    int*      ends   = (int*)(base + ends_off);
    float*    sumbuf = (float*)(base + sum_off);
    unsigned* maxbuf = (unsigned*)(base + max_off);
    float*    pooled = (float*)(base + pool_off);

    float* emb_out  = (float*)d_out;                       // [32,100,64]
    float* grid_out = emb_out + (size_t)NB * NP * NEMB;    // [32,100]
    float* acti_out = grid_out + (size_t)NB * NP;          // [32,100]

    // zero pooling accumulators every call (key 0 == -inf under ordered map)
    hipMemsetAsync(sumbuf, 0, (size_t)NB * (NP + 1) * NFC * 4, stream);
    hipMemsetAsync(maxbuf, 0, (size_t)NB * (NP + 1) * NFC * 4, stream);

    // 1) STFT -> comb
    stft_comb_kernel<<<550, 256, 0, stream>>>(x, comb);

    // 2) fused weights per scale (KSP layout)
    for (int s = 0; s < 3; ++s) {
        int total = 48 * KpadArr[s];
        weff_prep_kernel<<<(total + 255) / 256, 256, 0, stream>>>(
            ms_w1[s], ms_w2[s], ms_b1[s], ms_b2[s],
            (half_t*)(base + weff_off[s]), biasE + s * 64, ksArr[s], KpadArr[s]);
    }

    // 3) activity + adaptive grid + ends
    activity_grid_kernel<<<NB, 256, 0, stream>>>(
        comb, a_w1, a_b1, a_g, a_bt, a_m, a_v, a_w2, a_b2,
        grid_out, acti_out, ends);

    // 4) fused conv branches (WMMA) + two-level segment pooling
    dim3 cgrid((NT + 95) / 96, NB);
    conv_branch_kernel<9><<<cgrid, 192, 0, stream>>>(
        x, (const half_t*)(base + weff_off[0]), biasE + 0 * 64,
        ms_g[0], ms_bt[0], ms_m[0], ms_v[0], ends, sumbuf, maxbuf, 0 * NOC);
    conv_branch_kernel<25><<<cgrid, 192, 0, stream>>>(
        x, (const half_t*)(base + weff_off[1]), biasE + 1 * 64,
        ms_g[1], ms_bt[1], ms_m[1], ms_v[1], ends, sumbuf, maxbuf, 1 * NOC);
    conv_branch_kernel<49><<<cgrid, 192, 0, stream>>>(
        x, (const half_t*)(base + weff_off[2]), biasE + 2 * 64,
        ms_g[2], ms_bt[2], ms_m[2], ms_v[2], ends, sumbuf, maxbuf, 2 * NOC);

    // 5) finalize pooled
    pool_fin_kernel<<<(NB * NP * NFC + 255) / 256, 256, 0, stream>>>(
        sumbuf, maxbuf, ends, pooled);

    // 6) projection (WMMA) -> emb
    proj_wmma_kernel<<<100, 256, 0, stream>>>(pooled, proj_w, proj_b, emb_out);
}
